// Refinement_515396076339
// MI455X (gfx1250) — matrix operand order repaired and checked by reading, not compiled, not run on hardware
//
#include <hip/hip_runtime.h>

// ---------------------------------------------------------------------------
// CHGNet-style gated message passing layer for MI455X (gfx1250).
// All dense math runs on v_wmma_f32_16x16x32_bf16 (fp32 accumulate).
// Segment sums use hardware global FP32 atomics.
// ---------------------------------------------------------------------------

typedef __bf16 bf16_t;
typedef __attribute__((ext_vector_type(16))) __bf16 v16bf;
typedef __attribute__((ext_vector_type(8)))  float  v8f;

#define D      128
#define NB     7
#define NNODE  50000
#define EU_    200000
#define ED_    400000

// padded LDS leading dims (multiples of 8 so 16B vector loads stay aligned,
// and row stride mod 64 banks is 4 -> 16 lanes hit 16 distinct banks)
#define LDX 392   // bf16, 3*D gathered input
#define LDH 136   // bf16, D-wide hidden
#define LDF 132   // f32,  D-wide staging

__device__ __forceinline__ float sigmoidf_(float x) { return 1.0f / (1.0f + __expf(-x)); }
__device__ __forceinline__ float siluf_(float x)    { return x * sigmoidf_(x); }

__device__ __forceinline__ void atomic_fadd(float* p, float v) {
  __hip_atomic_fetch_add(p, v, __ATOMIC_RELAXED, __HIP_MEMORY_SCOPE_AGENT);
}

// One 16x16 output tile of  A[16,K] (bf16, LDS, row-major) @ W[K,128] (bf16,
// pre-packed fragment layout).  KT = K/32 WMMA steps.
// A-fragment per ISA 7.12.2 (16-bit 16x32): lane r(0..15)/half h: two
// contiguous 8-elt runs at k = kt*32 + h*8 and kt*32 + 16 + h*8.
// B pre-packed so each lane reads 32 contiguous bytes.
template <int KT>
__device__ __forceinline__ v8f gemm_tile(const bf16_t* sA, int lda,
                                         const bf16_t* Wp, int nt, int lane) {
  v8f acc = {0.f, 0.f, 0.f, 0.f, 0.f, 0.f, 0.f, 0.f};
  const int r = lane & 15, h = lane >> 4;
  const bf16_t* a0 = sA + r * lda + h * 8;
  const bf16_t* b0 = Wp + ((size_t)(nt * KT) * 32 + lane) * 16;
#pragma unroll
  for (int kt = 0; kt < KT; ++kt) {
    v16bf a, b;
    __builtin_memcpy(&a, a0 + kt * 32, 16);
    __builtin_memcpy(((char*)&a) + 16, a0 + kt * 32 + 16, 16);
    __builtin_memcpy(&b, b0 + (size_t)kt * 512, 32);
    acc = __builtin_amdgcn_wmma_f32_16x16x32_bf16(false, a, false, b,
                                                  (short)0, acc, false, false);
  }
  return acc;
}

// core/gate branch: act(LN(silu(x@W1)@W2)); core writes silu(.) into sNL,
// gate multiplies sigmoid(.) into sNL.
__device__ __forceinline__ void gated_branch(
    const bf16_t* sX, bf16_t* sH, float* sF, float* sNL,
    float* sRed, float* sMV,
    const bf16_t* pW1, const bf16_t* pW2,
    const float* gamma, const float* beta,
    bool is_gate, int tid, int lane, int wave) {
  // GEMM1: [16,384]@[384,128] -> silu -> bf16 sH
#pragma unroll
  for (int hnt = 0; hnt < 2; ++hnt) {
    int nt = wave + hnt * 4;
    v8f acc = gemm_tile<12>(sX, LDX, pW1, nt, lane);
    int c = lane & 15, h = lane >> 4;
#pragma unroll
    for (int v = 0; v < 8; ++v)
      sH[(h * 8 + v) * LDH + nt * 16 + c] = (bf16_t)siluf_(acc[v]);
  }
  __syncthreads();
  // GEMM2: [16,128]@[128,128] -> f32 sF
#pragma unroll
  for (int hnt = 0; hnt < 2; ++hnt) {
    int nt = wave + hnt * 4;
    v8f acc = gemm_tile<4>(sH, LDH, pW2, nt, lane);
    int c = lane & 15, h = lane >> 4;
#pragma unroll
    for (int v = 0; v < 8; ++v)
      sF[(h * 8 + v) * LDF + nt * 16 + c] = acc[v];
  }
  __syncthreads();
  // LayerNorm stats: 8 partial sums per row, then reduce
  {
    int r = tid >> 3, g = tid & 7;
    float s = 0.f, ss = 0.f;
    for (int c = g; c < D; c += 8) {
      float y = sF[r * LDF + c];
      s += y; ss += y * y;
    }
    sRed[(r * 8 + g) * 2 + 0] = s;
    sRed[(r * 8 + g) * 2 + 1] = ss;
  }
  __syncthreads();
  if (tid < 16) {
    float s = 0.f, ss = 0.f;
    for (int g = 0; g < 8; ++g) {
      s  += sRed[(tid * 8 + g) * 2 + 0];
      ss += sRed[(tid * 8 + g) * 2 + 1];
    }
    float m   = s * (1.0f / D);
    float var = ss * (1.0f / D) - m * m;
    sMV[tid * 2 + 0] = m;
    sMV[tid * 2 + 1] = rsqrtf(var + 1e-5f);
  }
  __syncthreads();
  {
    int c = tid;
#pragma unroll
    for (int r = 0; r < 16; ++r) {
      float y = (sF[r * LDF + c] - sMV[r * 2]) * sMV[r * 2 + 1] * gamma[c] + beta[c];
      if (!is_gate) sNL[r * LDF + c]  = siluf_(y);
      else          sNL[r * LDF + c] *= sigmoidf_(y);
    }
  }
  __syncthreads();
}

// ---------------------------------------------------------------------------
// Edge kernel: one block = 16 directed edges, 128 threads (4 waves).
// ---------------------------------------------------------------------------
__global__ __launch_bounds__(128) void edge_kernel(
    const float* __restrict__ node_feat, const float* __restrict__ edge_feat,
    const float* __restrict__ smooth_w,
    const int* __restrict__ src_idx, const int* __restrict__ tgt_idx,
    const int* __restrict__ d2u, const float* __restrict__ W_env,
    const bf16_t* __restrict__ pWc1, const bf16_t* __restrict__ pWc2,
    const bf16_t* __restrict__ pWg1, const bf16_t* __restrict__ pWg2,
    const bf16_t* __restrict__ pWe1, const bf16_t* __restrict__ pWe2,
    const float* __restrict__ ln_c_g, const float* __restrict__ ln_c_b,
    const float* __restrict__ ln_g_g, const float* __restrict__ ln_g_b,
    float* __restrict__ node_agg, float* __restrict__ esum,
    float* __restrict__ cnt) {
  __shared__ alignas(16) bf16_t sX[16 * LDX];
  __shared__ alignas(16) bf16_t sH[16 * LDH];
  __shared__ float sF[16 * LDF];
  __shared__ float sNL[16 * LDF];
  __shared__ float sSW[16 * LDF];
  __shared__ float sRed[16 * 8 * 2];
  __shared__ float sMV[16 * 2];
  __shared__ int sU[16], sS[16], sT[16];

  const int tid = threadIdx.x, lane = tid & 31, wave = tid >> 5;
  const int e0 = blockIdx.x * 16;   // ED % 16 == 0

  if (tid < 16) {
    sU[tid] = d2u[e0 + tid];
    sS[tid] = src_idx[e0 + tid];
    sT[tid] = tgt_idx[e0 + tid];
  }
  __syncthreads();

  // Gather x = [edge_feat | tgt | src] -> bf16 LDS; envelope sw = smooth@W_env
#pragma unroll 1
  for (int r = 0; r < 16; ++r) {
    const int u = sU[r], s = sS[r], t = sT[r];
    for (int c = tid; c < 3 * D; c += 128) {
      float v;
      if (c < D)          v = edge_feat[(size_t)u * D + c];
      else if (c < 2 * D) v = node_feat[(size_t)t * D + (c - D)];
      else                v = node_feat[(size_t)s * D + (c - 2 * D)];
      sX[r * LDX + c] = (bf16_t)v;
    }
    {
      int c = tid;
      float acc = 0.f;
#pragma unroll
      for (int j = 0; j < NB; ++j)
        acc += smooth_w[(size_t)u * NB + j] * W_env[j * D + c];
      sSW[r * LDF + c] = acc;
    }
  }
  __syncthreads();

  gated_branch(sX, sH, sF, sNL, sRed, sMV, pWc1, pWc2, ln_c_g, ln_c_b, false,
               tid, lane, wave);
  gated_branch(sX, sH, sF, sNL, sRed, sMV, pWg1, pWg2, ln_g_g, ln_g_b, true,
               tid, lane, wave);
  // sNL = nl = core * gate

  // smooth = nl * sw, segment-sum onto target nodes via FP atomics
  {
    int c = tid;
#pragma unroll 1
    for (int r = 0; r < 16; ++r)
      atomic_fadd(&node_agg[(size_t)sT[r] * D + c],
                  sNL[r * LDF + c] * sSW[r * LDF + c]);
  }

  // edge FFN: silu(nl @ We1) @ We2 -> esum[undirected] (+count)
  {
    int c = tid;
#pragma unroll
    for (int r = 0; r < 16; ++r) sH[r * LDH + c] = (bf16_t)sNL[r * LDF + c];
  }
  __syncthreads();
  bf16_t* sH2 = sX;  // reuse gather buffer
#pragma unroll
  for (int hnt = 0; hnt < 2; ++hnt) {
    int nt = wave + hnt * 4;
    v8f acc = gemm_tile<4>(sH, LDH, pWe1, nt, lane);
    int c = lane & 15, h = lane >> 4;
#pragma unroll
    for (int v = 0; v < 8; ++v)
      sH2[(h * 8 + v) * LDH + nt * 16 + c] = (bf16_t)siluf_(acc[v]);
  }
  __syncthreads();
#pragma unroll
  for (int hnt = 0; hnt < 2; ++hnt) {
    int nt = wave + hnt * 4;
    v8f acc = gemm_tile<4>(sH2, LDH, pWe2, nt, lane);
    int c = lane & 15, h = lane >> 4;
#pragma unroll
    for (int v = 0; v < 8; ++v) {
      int r = h * 8 + v;
      atomic_fadd(&esum[(size_t)sU[r] * D + nt * 16 + c], acc[v]);
    }
  }
  if (tid < 16) atomic_fadd(&cnt[sU[tid]], 1.0f);
}

// ---------------------------------------------------------------------------
// Node kernel: 16 nodes / block; silu(node_agg@Wn1)@Wn2 + res*node_feat
// ---------------------------------------------------------------------------
__global__ __launch_bounds__(128) void node_kernel(
    const float* __restrict__ node_agg, const float* __restrict__ node_feat,
    const bf16_t* __restrict__ pWn1, const bf16_t* __restrict__ pWn2,
    const float* __restrict__ node_res_w, float* __restrict__ out_node) {
  __shared__ alignas(16) bf16_t sA[16 * LDH];
  __shared__ alignas(16) bf16_t sB[16 * LDH];
  const int tid = threadIdx.x, lane = tid & 31, wave = tid >> 5;
  const int n0 = blockIdx.x * 16;  // N % 16 == 0
  {
    int c = tid;
#pragma unroll
    for (int r = 0; r < 16; ++r)
      sA[r * LDH + c] = (bf16_t)node_agg[(size_t)(n0 + r) * D + c];
  }
  __syncthreads();
#pragma unroll
  for (int hnt = 0; hnt < 2; ++hnt) {
    int nt = wave + hnt * 4;
    v8f acc = gemm_tile<4>(sA, LDH, pWn1, nt, lane);
    int c = lane & 15, h = lane >> 4;
#pragma unroll
    for (int v = 0; v < 8; ++v)
      sB[(h * 8 + v) * LDH + nt * 16 + c] = (bf16_t)siluf_(acc[v]);
  }
  __syncthreads();
#pragma unroll
  for (int hnt = 0; hnt < 2; ++hnt) {
    int nt = wave + hnt * 4;
    v8f acc = gemm_tile<4>(sB, LDH, pWn2, nt, lane);
    int c = lane & 15, h = lane >> 4;
#pragma unroll
    for (int v = 0; v < 8; ++v) {
      int r = n0 + h * 8 + v;
      int cc = nt * 16 + c;
      out_node[(size_t)r * D + cc] =
          acc[v] + node_res_w[cc] * node_feat[(size_t)r * D + cc];
    }
  }
}

// ---------------------------------------------------------------------------
// Edge finalize: delta_edge = esum / max(cnt,1) + res*edge_feat
// ---------------------------------------------------------------------------
__global__ void edge_final(const float* __restrict__ esum,
                           const float* __restrict__ cnt,
                           const float* __restrict__ edge_feat,
                           const float* __restrict__ edge_res_w,
                           float* __restrict__ out_edge) {
  size_t i = (size_t)blockIdx.x * blockDim.x + threadIdx.x;
  if (i >= (size_t)EU_ * D) return;
  int u = (int)(i >> 7), c = (int)(i & 127);
  float ct = cnt[u];
  ct = ct < 1.f ? 1.f : ct;
  out_edge[i] = esum[i] / ct + edge_res_w[c] * edge_feat[i];
}

// ---------------------------------------------------------------------------
// Weight pack: fp32 [K,128] row-major -> bf16 B-fragment layout
// idx = ((nt*KT + kt)*32 + lane)*16 + i ; k = kt*32 + 16*(i/8) + 8*(lane/16) + i%8
// ---------------------------------------------------------------------------
__global__ void pack_weights(const float* __restrict__ W,
                             bf16_t* __restrict__ out, int KT) {
  int id = blockIdx.x * blockDim.x + threadIdx.x;
  int total = KT * 8 * 512;
  if (id >= total) return;
  int i = id & 15;
  int lane = (id >> 4) & 31;
  int kt = (id >> 9) % KT;
  int nt = id / (KT << 9);
  int p = i >> 1, lo = i & 1;
  int k = kt * 32 + ((p >> 2) << 4) + ((lane >> 4) << 3) + ((p & 3) << 1) + lo;
  int c = nt * 16 + (lane & 15);
  out[id] = (bf16_t)W[(size_t)k * D + c];
}

extern "C" void kernel_launch(void* const* d_in, const int* in_sizes, int n_in,
                              void* d_out, int out_size, void* d_ws,
                              size_t ws_size, hipStream_t stream) {
  const float* node_feat = (const float*)d_in[0];
  const float* edge_feat = (const float*)d_in[1];
  const float* smooth_w  = (const float*)d_in[2];
  const int*   src_idx   = (const int*)d_in[3];
  const int*   tgt_idx   = (const int*)d_in[4];
  const int*   d2u       = (const int*)d_in[5];
  const float* W_env     = (const float*)d_in[6];
  const float* Wc1       = (const float*)d_in[7];
  const float* Wc2       = (const float*)d_in[8];
  const float* Wg1       = (const float*)d_in[9];
  const float* Wg2       = (const float*)d_in[10];
  const float* ln_c_g    = (const float*)d_in[11];
  const float* ln_c_b    = (const float*)d_in[12];
  const float* ln_g_g    = (const float*)d_in[13];
  const float* ln_g_b    = (const float*)d_in[14];
  const float* Wn1       = (const float*)d_in[15];
  const float* Wn2       = (const float*)d_in[16];
  const float* We1       = (const float*)d_in[17];
  const float* We2       = (const float*)d_in[18];
  const float* node_res_w = (const float*)d_in[19];
  const float* edge_res_w = (const float*)d_in[20];

  // workspace layout
  float* node_agg = (float*)d_ws;                       // N*D
  float* esum = node_agg + (size_t)NNODE * D;           // EU*D
  float* cnt  = esum + (size_t)EU_ * D;                 // EU
  bf16_t* pk  = (bf16_t*)(cnt + EU_);                   // packed weights
  const size_t P384 = (size_t)12 * 8 * 512;             // 49152 bf16
  const size_t P128 = (size_t)4 * 8 * 512;              // 16384 bf16
  bf16_t* pWc1 = pk;
  bf16_t* pWg1 = pWc1 + P384;
  bf16_t* pWc2 = pWg1 + P384;
  bf16_t* pWg2 = pWc2 + P128;
  bf16_t* pWn1 = pWg2 + P128;
  bf16_t* pWn2 = pWn1 + P128;
  bf16_t* pWe1 = pWn2 + P128;
  bf16_t* pWe2 = pWe1 + P128;
  size_t zero_bytes = ((size_t)NNODE * D + (size_t)EU_ * D + EU_) * sizeof(float);
  size_t need = zero_bytes + (2 * P384 + 6 * P128) * sizeof(bf16_t);
  if (ws_size < need) return;

  hipMemsetAsync(d_ws, 0, zero_bytes, stream);

  pack_weights<<<(int)((P384 + 255) / 256), 256, 0, stream>>>(Wc1, pWc1, 12);
  pack_weights<<<(int)((P384 + 255) / 256), 256, 0, stream>>>(Wg1, pWg1, 12);
  pack_weights<<<(int)((P128 + 255) / 256), 256, 0, stream>>>(Wc2, pWc2, 4);
  pack_weights<<<(int)((P128 + 255) / 256), 256, 0, stream>>>(Wg2, pWg2, 4);
  pack_weights<<<(int)((P128 + 255) / 256), 256, 0, stream>>>(Wn1, pWn1, 4);
  pack_weights<<<(int)((P128 + 255) / 256), 256, 0, stream>>>(Wn2, pWn2, 4);
  pack_weights<<<(int)((P128 + 255) / 256), 256, 0, stream>>>(We1, pWe1, 4);
  pack_weights<<<(int)((P128 + 255) / 256), 256, 0, stream>>>(We2, pWe2, 4);

  edge_kernel<<<ED_ / 16, 128, 0, stream>>>(
      node_feat, edge_feat, smooth_w, src_idx, tgt_idx, d2u, W_env,
      pWc1, pWc2, pWg1, pWg2, pWe1, pWe2,
      ln_c_g, ln_c_b, ln_g_g, ln_g_b, node_agg, esum, cnt);

  float* out_node = (float*)d_out;
  float* out_edge = out_node + (size_t)NNODE * D;
  node_kernel<<<NNODE / 16, 128, 0, stream>>>(node_agg, node_feat, pWn1, pWn2,
                                              node_res_w, out_node);
  edge_final<<<(int)(((size_t)EU_ * D + 255) / 256), 256, 0, stream>>>(
      esum, cnt, edge_feat, edge_res_w, out_edge);
}
